// TransformerBlock_18794776887333
// MI455X (gfx1250) — matrix operand order repaired and checked by reading, not compiled, not run on hardware
//
#include <hip/hip_runtime.h>
#include <hip/hip_bf16.h>

// ---------------------------------------------------------------------------
// Problem constants (fixed by the reference): B=2, L=2048, D=2048, H=16,
// DH=128, FF=8192.  BL = 4096.  All GEMM dims are multiples of the 128x128
// block tile and the K-step of 32, so no edge handling is required.
// ---------------------------------------------------------------------------
#define BATCH   2
#define SEQ_L   2048
#define DMODEL  2048
#define NHEAD   16
#define DHEAD   128
#define FFDIM   8192
#define BLROWS  (BATCH * SEQ_L)     // 4096
#define NEGINF  (-1.0e9f)

typedef __attribute__((ext_vector_type(16))) __bf16 v16bf;
typedef __attribute__((ext_vector_type(8)))  float  v8f;

union Frag { v16bf v; uint4 q[2]; };

__device__ __forceinline__ v8f zero8() {
    v8f z;
#pragma unroll
    for (int i = 0; i < 8; i++) z[i] = 0.0f;
    return z;
}

__device__ __forceinline__ v8f wmma_bf16(v16bf a, v16bf b, v8f c) {
    // D = A(16x32 bf16) * B(32x16 bf16) + C(16x16 f32)
    return __builtin_amdgcn_wmma_f32_16x16x32_bf16(
        /*neg_a=*/false, a, /*neg_b=*/false, b,
        /*c_mod=*/(short)0, c, /*reuse_a=*/false, /*reuse_b=*/false);
}

// ---------------------------------------------------------------------------
// Tensor Data Mover support (gfx1250).  One TDM descriptor DMAs a full
// 128x32 bf16 tile (row stride K elements) from global memory into LDS,
// inserting 16B of padding after every 64B row (pad_interval = 16 DWORDs,
// pad_amount = 4 DWORDs) to match the bank-conflict-free LDS layout.
// ---------------------------------------------------------------------------
#if defined(__has_builtin)
#if __has_builtin(__builtin_amdgcn_tensor_load_to_lds)
#define USE_TDM 1
#endif
#endif
#ifndef USE_TDM
#define USE_TDM 0
#endif

#if USE_TDM
typedef unsigned int tdm_u32x4 __attribute__((ext_vector_type(4)));
typedef int          tdm_i32x4 __attribute__((ext_vector_type(4)));
typedef int          tdm_i32x8 __attribute__((ext_vector_type(8)));

__device__ __forceinline__ void tdm_load_tile_128x32(
    const __bf16* gtile, unsigned lds_byte_addr, unsigned Kelems)
{
    const unsigned long long ga = (unsigned long long)(uintptr_t)gtile;
    tdm_u32x4 g0;
    g0[0] = 1u;                                    // count=1, user descriptor
    g0[1] = lds_byte_addr;                         // LDS destination (bytes)
    g0[2] = (unsigned)ga;                          // global_addr[31:0]
    g0[3] = (unsigned)((ga >> 32) & 0x01FFFFFFull) // global_addr[56:32]
          | (2u << 30);                            // type = 2 ("image")
    tdm_i32x8 g1;
    g1[0] = (int)((1u << 16)                       // data_size = 2 bytes
                | (1u << 20)                       // pad_enable
                | (3u << 22)                       // pad_interval = 16 DWORDs (64B)
                | (3u << 25));                     // pad_amount   = 4 DWORDs (16B)
    g1[1] = (int)((Kelems & 0xFFFFu) << 16);       // tensor_dim0[15:0]
    g1[2] = (int)((Kelems >> 16) | (0xFFFFu << 16)); // dim0[31:16] | tensor_dim1[15:0]=65535
    g1[3] = (int)(32u << 16);                      // tensor_dim1[31:16]=0 | tile_dim0 = 32
    g1[4] = (int)128u;                             // tile_dim1 = 128, tile_dim2 = 0
    g1[5] = (int)Kelems;                           // tensor_dim0_stride[31:0] = K
    g1[6] = 0;                                     // stride0[47:32]=0, stride1[15:0]=0
    g1[7] = 0;
    tdm_i32x4 z4 = (tdm_i32x4)0;
#if __clang_major__ >= 23
    tdm_i32x8 z8 = (tdm_i32x8)0;
    __builtin_amdgcn_tensor_load_to_lds(g0, g1, z4, z4, z8, 0);
#else
    __builtin_amdgcn_tensor_load_to_lds(g0, g1, z4, z4, 0);
#endif
}
#endif // USE_TDM

// ---------------------------------------------------------------------------
// fp32 -> bf16 conversion (vectorized x4)
// ---------------------------------------------------------------------------
__global__ void cvt_f32_bf16(const float* __restrict__ in,
                             __bf16* __restrict__ out, int n) {
    int i   = blockIdx.x * blockDim.x + threadIdx.x;
    int idx = i * 4;
    if (idx + 3 < n) {
        float4 f = *(const float4*)(in + idx);
        out[idx + 0] = (__bf16)f.x;
        out[idx + 1] = (__bf16)f.y;
        out[idx + 2] = (__bf16)f.z;
        out[idx + 3] = (__bf16)f.w;
    } else {
        for (int k = idx; k < n; k++) out[k] = (__bf16)in[k];
    }
}

// ---------------------------------------------------------------------------
// Generic bf16 NT GEMM:  C[M,N] = A[M,K] * Bw[N,K]^T  (+bias)(+resid)(relu)
// Block tile 128x128, 8 waves of 64x32, K-step 32, double-buffered LDS:
// tile k+1 is staged (TDM DMA or manual b128 loads) while tile k feeds WMMA,
// with a single workgroup barrier per K-step.
// out_mode: 0 = [M,N], 1 = head-split [B,H,L,DH], 2 = head-transposed
//           [B,H,DH,L] (used for V so P*V B-operand is K-contiguous).
// ---------------------------------------------------------------------------
#define TM 128
#define TN 128
#define TKS 32
#define LDS_STRIDE 80   // 64B of bf16 data + 16B pad, keeps b128 LDS loads 16B aligned
#define BUFBYTES ((TM + TN) * LDS_STRIDE)   // 20480B per double-buffer stage

__global__ __launch_bounds__(256) void gemm_bf16_nt(
    const __bf16* __restrict__ A, const __bf16* __restrict__ Bw,
    const float* __restrict__ bias, const float* __restrict__ resid,
    float* __restrict__ outF, __bf16* __restrict__ outB,
    int M, int N, int K, int relu, int out_mode)
{
    __shared__ uint4 smem4[2 * BUFBYTES / 16];
    char* base = (char*)smem4;

    const int tid  = threadIdx.x;
    const int lane = tid & 31, wid = tid >> 5;
    const int lo   = lane & 15, hi = lane >> 4;
    const int wm   = wid & 1;          // 2 wave-rows of 64
    const int wn   = wid >> 1;         // 4 wave-cols of 32
    const int mBase = blockIdx.y * TM;
    const int nBase = blockIdx.x * TN;
    const int nk = K / TKS;

    v8f acc[4][2];
#pragma unroll
    for (int i = 0; i < 4; i++)
#pragma unroll
        for (int j = 0; j < 2; j++) acc[i][j] = zero8();

#if USE_TDM
    const unsigned ldsBase = (unsigned)(uintptr_t)base;  // low 32 bits = LDS offset
    // stage tile 0 into buffer 0
    if (wid == 0) {
        tdm_load_tile_128x32(A  + (size_t)mBase * K, ldsBase,                    (unsigned)K);
        tdm_load_tile_128x32(Bw + (size_t)nBase * K, ldsBase + TM * LDS_STRIDE,  (unsigned)K);
        __builtin_amdgcn_s_wait_tensorcnt(0);
    }
#else
    const int lr   = tid >> 1;   // tile row loaded by this thread
    const int lseg = tid & 1;    // which 32B half of the 64B row
    {   // stage tile 0 into buffer 0
        const __bf16* gA = A  + (size_t)(mBase + lr) * K + lseg * 16;
        const __bf16* gB = Bw + (size_t)(nBase + lr) * K + lseg * 16;
        uint4 a0 = ((const uint4*)gA)[0], a1 = ((const uint4*)gA)[1];
        uint4 b0 = ((const uint4*)gB)[0], b1 = ((const uint4*)gB)[1];
        uint4* dA = (uint4*)(base + lr * LDS_STRIDE + lseg * 32);
        uint4* dB = (uint4*)(base + TM * LDS_STRIDE + lr * LDS_STRIDE + lseg * 32);
        dA[0] = a0; dA[1] = a1;
        dB[0] = b0; dB[1] = b1;
    }
#endif

    int cur = 0;
    for (int kt = 0; kt < nk; kt++) {
        __syncthreads();                 // buffer `cur` is ready; `nxt` is free
        const int nxt = cur ^ 1;
        const int kbn = (kt + 1) * TKS;

#if USE_TDM
        if (wid == 0 && kt + 1 < nk) {   // DMA next tile while this one computes
            const unsigned off = ldsBase + nxt * BUFBYTES;
            tdm_load_tile_128x32(A  + (size_t)mBase * K + kbn, off,                   (unsigned)K);
            tdm_load_tile_128x32(Bw + (size_t)nBase * K + kbn, off + TM * LDS_STRIDE, (unsigned)K);
        }
#else
        uint4 a0, a1, b0, b1;
        if (kt + 1 < nk) {               // global loads overlap the WMMA work
            const __bf16* gA = A  + (size_t)(mBase + lr) * K + kbn + lseg * 16;
            const __bf16* gB = Bw + (size_t)(nBase + lr) * K + kbn + lseg * 16;
            __builtin_prefetch(gA + TKS, 0, 1);
            __builtin_prefetch(gB + TKS, 0, 1);
            a0 = ((const uint4*)gA)[0]; a1 = ((const uint4*)gA)[1];
            b0 = ((const uint4*)gB)[0]; b1 = ((const uint4*)gB)[1];
        }
#endif

        char* sA = base + cur * BUFBYTES;
        char* sB = sA + TM * LDS_STRIDE;

        // ---- B fragments: lane lo = column N, K contiguous per 16-lane half
        Frag bf[2];
#pragma unroll
        for (int ni = 0; ni < 2; ni++) {
            const char* p = sB + (wn * 32 + ni * 16 + lo) * LDS_STRIDE + hi * 32;
            bf[ni].q[0] = *(const uint4*)p;
            bf[ni].q[1] = *(const uint4*)(p + 16);
        }
        // ---- A fragments (K contiguous per 16-lane half) + WMMA
#pragma unroll
        for (int mi = 0; mi < 4; mi++) {
            Frag af;
            const char* p = sA + (wm * 64 + mi * 16 + lo) * LDS_STRIDE + hi * 32;
            af.q[0] = *(const uint4*)p;
            af.q[1] = *(const uint4*)(p + 16);
#pragma unroll
            for (int ni = 0; ni < 2; ni++)
                acc[mi][ni] = wmma_bf16(af.v, bf[ni].v, acc[mi][ni]);
        }

#if USE_TDM
        if (wid == 0 && kt + 1 < nk)
            __builtin_amdgcn_s_wait_tensorcnt(0);   // next tile landed in LDS
#else
        if (kt + 1 < nk) {
            uint4* dA = (uint4*)(base + nxt * BUFBYTES + lr * LDS_STRIDE + lseg * 32);
            uint4* dB = (uint4*)(base + nxt * BUFBYTES + TM * LDS_STRIDE + lr * LDS_STRIDE + lseg * 32);
            dA[0] = a0; dA[1] = a1;
            dB[0] = b0; dB[1] = b1;
        }
#endif
        cur = nxt;
    }

    // ---- epilogue: C layout is (reg r, lanes0-15 -> M=r, lanes16-31 -> M=r+8)
#pragma unroll
    for (int mi = 0; mi < 4; mi++) {
#pragma unroll
        for (int ni = 0; ni < 2; ni++) {
#pragma unroll
            for (int r = 0; r < 8; r++) {
                const int m = mBase + wm * 64 + mi * 16 + r + hi * 8;
                const int n = nBase + wn * 32 + ni * 16 + lo;
                float v = acc[mi][ni][r];
                if (bias)  v += bias[n];
                if (resid) v += resid[(size_t)m * N + n];
                if (relu)  v = v > 0.0f ? v : 0.0f;
                if (outF)  outF[(size_t)m * N + n] = v;
                if (outB) {
                    size_t idx;
                    if (out_mode == 0) {
                        idx = (size_t)m * N + n;
                    } else {
                        const int b  = m / SEQ_L, l = m % SEQ_L;
                        const int h  = n / DHEAD, dh = n % DHEAD;
                        if (out_mode == 1)
                            idx = (((size_t)b * NHEAD + h) * SEQ_L + l) * DHEAD + dh;
                        else
                            idx = (((size_t)b * NHEAD + h) * DHEAD + dh) * SEQ_L + l;
                    }
                    outB[idx] = (__bf16)v;
                }
            }
        }
    }
}

// ---------------------------------------------------------------------------
// Flash attention (causal), bf16 WMMA, fp32 online softmax.
// Grid: (L/128, B*H). Block: 256 threads = 8 waves, 16 queries per wave.
// Q/K layout [B,H,L,DH]; V stored transposed [B,H,DH,L]; output [B,L,D] bf16.
// ---------------------------------------------------------------------------
#define KSTR 272   // 256B of K-row data + 16B pad
#define VSTR 80    // 64B  of V^T-row data + 16B pad
#define PSTR 80

__global__ __launch_bounds__(256) void attn_flash(
    const __bf16* __restrict__ Q, const __bf16* __restrict__ Kc,
    const __bf16* __restrict__ Vt, __bf16* __restrict__ O)
{
    __shared__ uint4 smem4[(32 * KSTR + 128 * VSTR + 8 * 16 * PSTR) / 16];
    char* sK = (char*)smem4;
    char* sV = sK + 32 * KSTR;
    char* sP = sV + 128 * VSTR;

    const int tid  = threadIdx.x;
    const int lane = tid & 31, wid = tid >> 5;
    const int lo   = lane & 15, hi = lane >> 4;
    const int bh    = blockIdx.y;          // b*H + h
    const int qBase = blockIdx.x * 128;
    const int q0    = qBase + wid * 16;    // this wave's 16 queries

    // Q fragments (16 x 128 bf16 = 4 A-fragments), kept in registers
    Frag qf[4];
    {
        const char* qrow = (const char*)(Q + ((size_t)bh * SEQ_L + q0 + lo) * DHEAD);
#pragma unroll
        for (int c = 0; c < 4; c++) {
            const char* p = qrow + c * 64 + hi * 32;
            qf[c].q[0] = *(const uint4*)p;
            qf[c].q[1] = *(const uint4*)(p + 16);
        }
    }

    v8f   o[8];
    float mrun[8], lrun[8];
#pragma unroll
    for (int t = 0; t < 8; t++) o[t] = zero8();
#pragma unroll
    for (int r = 0; r < 8; r++) { mrun[r] = -1.0e30f; lrun[r] = 0.0f; }

    const int nkb = (qBase + 128) >> 5;    // causal: keys up to block's max query
    for (int kt = 0; kt < nkb; kt++) {
        const int kBase = kt * 32;
        __syncthreads();
        // ---- K tile: 32 keys x 128 dh ----
        {
            const int r = tid >> 3, seg = tid & 7;
            const __bf16* g = Kc + ((size_t)bh * SEQ_L + kBase + r) * DHEAD + seg * 16;
            uint4 a0 = ((const uint4*)g)[0];
            uint4 a1 = ((const uint4*)g)[1];
            uint4* d = (uint4*)(sK + r * KSTR + seg * 32);
            d[0] = a0; d[1] = a1;
        }
        // ---- V^T tile: 128 dh x 32 keys ----
        {
            const int r = tid >> 1, seg = tid & 1;
            const __bf16* g = Vt + ((size_t)bh * DHEAD + r) * SEQ_L + kBase + seg * 16;
            uint4 a0 = ((const uint4*)g)[0];
            uint4 a1 = ((const uint4*)g)[1];
            uint4* d = (uint4*)(sV + r * VSTR + seg * 32);
            d[0] = a0; d[1] = a1;
        }
        __syncthreads();

        // ---- S = Q K^T  (16 x 32, 8 WMMAs over dh) ----
        v8f s[2] = { zero8(), zero8() };
#pragma unroll
        for (int c = 0; c < 4; c++) {
#pragma unroll
            for (int ni = 0; ni < 2; ni++) {
                Frag kf;   // B-operand: lane lo = key, dh contiguous per half
                const char* p = sK + (ni * 16 + lo) * KSTR + c * 64 + hi * 32;
                kf.q[0] = *(const uint4*)p;
                kf.q[1] = *(const uint4*)(p + 16);
                s[ni] = wmma_bf16(qf[c].v, kf.v, s[ni]);
            }
        }

        // ---- causal mask + online softmax (rows live in 16-lane groups) ----
        float sc[8];
#pragma unroll
        for (int r = 0; r < 8; r++) {
            const int q = q0 + r + hi * 8;
#pragma unroll
            for (int ni = 0; ni < 2; ni++) {
                const int k = kBase + ni * 16 + lo;
                if (k > q) s[ni][r] += NEGINF;
            }
            float mx = fmaxf(s[0][r], s[1][r]);
            mx = fmaxf(mx, __shfl_xor(mx, 1));
            mx = fmaxf(mx, __shfl_xor(mx, 2));
            mx = fmaxf(mx, __shfl_xor(mx, 4));
            mx = fmaxf(mx, __shfl_xor(mx, 8));
            const float mnew = fmaxf(mrun[r], mx);
            sc[r] = __expf(mrun[r] - mnew);
            const float p0 = __expf(s[0][r] - mnew);
            const float p1 = __expf(s[1][r] - mnew);
            s[0][r] = p0; s[1][r] = p1;
            float rs = p0 + p1;
            rs += __shfl_xor(rs, 1);
            rs += __shfl_xor(rs, 2);
            rs += __shfl_xor(rs, 4);
            rs += __shfl_xor(rs, 8);
            lrun[r] = lrun[r] * sc[r] + rs;
            mrun[r] = mnew;
        }
        // rescale accumulated O
#pragma unroll
        for (int t = 0; t < 8; t++)
#pragma unroll
            for (int r = 0; r < 8; r++) o[t][r] *= sc[r];

        // ---- repack P (16x32 bf16) through per-wave LDS into A layout ----
        char* pw = sP + wid * 16 * PSTR;
#pragma unroll
        for (int ni = 0; ni < 2; ni++)
#pragma unroll
            for (int r = 0; r < 8; r++)
                ((__bf16*)(pw + (r + hi * 8) * PSTR))[ni * 16 + lo] = (__bf16)s[ni][r];

        Frag pf;
        {
            const char* p = pw + lo * PSTR + hi * 32;
            pf.q[0] = *(const uint4*)p;
            pf.q[1] = *(const uint4*)(p + 16);
        }
        // ---- O += P * V  (8 WMMAs over dh columns) ----
#pragma unroll
        for (int nt = 0; nt < 8; nt++) {
            Frag vf;   // B-operand: lane lo = dh, keys contiguous per half
            const char* p = sV + (nt * 16 + lo) * VSTR + hi * 32;
            vf.q[0] = *(const uint4*)p;
            vf.q[1] = *(const uint4*)(p + 16);
            o[nt] = wmma_bf16(pf.v, vf.v, o[nt]);
        }
    }

    // ---- normalize + merge heads: out[b, q, h*DH + dh] (bf16) ----
    const int b = bh / NHEAD, h = bh % NHEAD;
#pragma unroll
    for (int nt = 0; nt < 8; nt++) {
#pragma unroll
        for (int r = 0; r < 8; r++) {
            const int q  = q0 + r + hi * 8;
            const int dh = nt * 16 + lo;
            const float v = o[nt][r] / lrun[r];
            O[((size_t)b * SEQ_L + q) * DMODEL + h * DHEAD + dh] = (__bf16)v;
        }
    }
}

// ---------------------------------------------------------------------------
// Host-side orchestration
// ---------------------------------------------------------------------------
extern "C" void kernel_launch(void* const* d_in, const int* in_sizes, int n_in,
                              void* d_out, int out_size, void* d_ws, size_t ws_size,
                              hipStream_t stream) {
    (void)in_sizes; (void)n_in; (void)out_size; (void)ws_size;
    const float* x     = (const float*)d_in[0];
    // d_in[1] = mask : unused, causal mask is computed analytically
    const float* Wq    = (const float*)d_in[2];
    const float* Wk    = (const float*)d_in[3];
    const float* Wv    = (const float*)d_in[4];
    const float* Wo    = (const float*)d_in[5];
    const float* Wup   = (const float*)d_in[6];
    const float* bup   = (const float*)d_in[7];
    const float* Wdown = (const float*)d_in[8];

    char* ws = (char*)d_ws;
    const size_t SZ_X16  = (size_t)BLROWS * DMODEL * 2;   // 16 MB bf16 [BL,D]
    const size_t SZ_W16  = (size_t)DMODEL * DMODEL * 2;   //  8 MB bf16 [D,D]
    const size_t SZ_FF16 = (size_t)FFDIM  * DMODEL * 2;   // 32 MB bf16 [FF,D]
    const size_t SZ_X32  = (size_t)BLROWS * DMODEL * 4;   // 32 MB f32  [BL,D]
    const size_t SZ_H16  = (size_t)BLROWS * FFDIM  * 2;   // 64 MB bf16 [BL,FF]

    size_t off = 0;
    __bf16* xb    = (__bf16*)(ws + off); off += SZ_X16;
    __bf16* qb    = (__bf16*)(ws + off); off += SZ_X16;
    __bf16* kb    = (__bf16*)(ws + off); off += SZ_X16;
    __bf16* vb    = (__bf16*)(ws + off); off += SZ_X16;   // transposed [B,H,DH,L]
    __bf16* attnb = (__bf16*)(ws + off); off += SZ_X16;
    __bf16* x2b   = (__bf16*)(ws + off); off += SZ_X16;
    __bf16* hb    = (__bf16*)(ws + off); off += SZ_H16;
    float*  x2f   = (float* )(ws + off); off += SZ_X32;
    __bf16* wqb   = (__bf16*)(ws + off); off += SZ_W16;
    __bf16* wkb   = (__bf16*)(ws + off); off += SZ_W16;
    __bf16* wvb   = (__bf16*)(ws + off); off += SZ_W16;
    __bf16* wob   = (__bf16*)(ws + off); off += SZ_W16;
    __bf16* wupb  = (__bf16*)(ws + off); off += SZ_FF16;
    __bf16* wdnb  = (__bf16*)(ws + off); off += SZ_FF16;

    auto cvt = [&](const float* src, __bf16* dst, int n) {
        int blocks = (n / 4 + 255) / 256;
        cvt_f32_bf16<<<blocks, 256, 0, stream>>>(src, dst, n);
    };
    cvt(x,     xb,   BLROWS * DMODEL);
    cvt(Wq,    wqb,  DMODEL * DMODEL);
    cvt(Wk,    wkb,  DMODEL * DMODEL);
    cvt(Wv,    wvb,  DMODEL * DMODEL);
    cvt(Wo,    wob,  DMODEL * DMODEL);
    cvt(Wup,   wupb, FFDIM  * DMODEL);
    cvt(Wdown, wdnb, DMODEL * FFDIM);

    const dim3 blk(256);
    const dim3 grid_d (DMODEL / TN, BLROWS / TM);   // (16, 32)
    const dim3 grid_ff(FFDIM  / TN, BLROWS / TM);   // (64, 32)

    // Q, K, V projections (head-split; V transposed for P*V)
    gemm_bf16_nt<<<grid_d, blk, 0, stream>>>(xb, wqb, nullptr, nullptr,
        nullptr, qb, BLROWS, DMODEL, DMODEL, 0, 1);
    gemm_bf16_nt<<<grid_d, blk, 0, stream>>>(xb, wkb, nullptr, nullptr,
        nullptr, kb, BLROWS, DMODEL, DMODEL, 0, 1);
    gemm_bf16_nt<<<grid_d, blk, 0, stream>>>(xb, wvb, nullptr, nullptr,
        nullptr, vb, BLROWS, DMODEL, DMODEL, 0, 2);

    // causal flash attention
    const dim3 agrid(SEQ_L / 128, BATCH * NHEAD);   // (16, 32)
    attn_flash<<<agrid, blk, 0, stream>>>(qb, kb, vb, attnb);

    // x2 = x + attn @ Wo^T   (f32 for final residual + bf16 for FFN input)
    gemm_bf16_nt<<<grid_d, blk, 0, stream>>>(attnb, wob, nullptr, x,
        x2f, x2b, BLROWS, DMODEL, DMODEL, 0, 0);

    // h = relu(x2 @ W_up^T + b_up)
    gemm_bf16_nt<<<grid_ff, blk, 0, stream>>>(x2b, wupb, bup, nullptr,
        nullptr, hb, BLROWS, FFDIM, DMODEL, 1, 0);

    // out = x2 + h @ W_down^T   (f32 -> d_out)
    gemm_bf16_nt<<<grid_d, blk, 0, stream>>>(hb, wdnb, nullptr, x2f,
        (float*)d_out, nullptr, BLROWS, DMODEL, FFDIM, 0, 0);
}
